// WindowAttention_83640193122718
// MI455X (gfx1250) — matrix operand order repaired and checked by reading, not compiled, not run on hardware
//
#include <hip/hip_runtime.h>
#include <hip/hip_bf16.h>
#include <cstddef>

// ---------------------------------------------------------------------------
// Window attention, fused per-window kernel for gfx1250 (MI455X, wave32, WMMA)
//   B_=2048 windows, N=49 tokens, C=384, H=12 heads, d=32, nW=64 masks
// All GEMMs mapped to v_wmma_f32_16x16x32_bf16; operands staged so every
// fragment is a contiguous per-lane 16B/32B run (-> ds_load_b128 /
// global_load_b128), weights pre-converted to bf16 once into d_ws.
// ---------------------------------------------------------------------------

#define N_TOK   49
#define DIM     384
#define NHEADS  12
#define HEADD   32
#define NWIN    2048
#define NMASK   64
#define QSCALE  0.17677669529663689f   // 32^-0.5

typedef __attribute__((ext_vector_type(16))) __bf16 v16bf;
typedef __attribute__((ext_vector_type(8)))  float  v8f;

union Frag { v16bf v; unsigned int u[8]; uint4 q[2]; };

// round-to-nearest-even f32 -> bf16 (bit trick)
__device__ __forceinline__ unsigned short f2bf(float f) {
    unsigned int u = __float_as_uint(f);
    u += 0x7fffu + ((u >> 16) & 1u);
    return (unsigned short)(u >> 16);
}
__device__ __forceinline__ unsigned int pack2(float lo, float hi) {
    return (unsigned int)f2bf(lo) | ((unsigned int)f2bf(hi) << 16);
}

// Dynamic LDS layout (bytes); all tiles padded to 64 rows so fragment loads
// are unconditional b128s:
//   xb  [64][384] bf16 : 49152   window input (rows 49..63 zeroed)
//   qh  [2][64][32]    :  8192   q (row-major, pre-scaled), per head pair
//   kh  [2][64][32]    :  8192   k (row-major == B^T for QK^T)
//   vT  [2][32][64]    :  8192   v transposed (== B^T for AV)
//   at  [8][16][64]    : 16384   per-wave softmaxed probs (A-frag staging)
//   os  [64][384]      : 49152   attention output (rows 49..63 zeroed)
#define XB_OFF 0
#define QH_OFF 49152
#define KH_OFF 57344
#define VT_OFF 65536
#define AT_OFF 73728
#define OS_OFF 90112
#define SMEM_BYTES 139264

// ws layout (bytes): posb f32 [12*49*49] @0 ; qkv_w bf16 @115248 ; proj_w bf16 @999984
#define WS_QKVW_OFF 115248
#define WS_PROJW_OFF 999984

// ---------------------------------------------------------------------------
__global__ void posbias_kernel(const float* __restrict__ w1, const float* __restrict__ b1,
                               const float* __restrict__ w2, const float* __restrict__ b2,
                               float* __restrict__ bias) {
    int idx = blockIdx.x * 256 + threadIdx.x;
    if (idx >= NHEADS * N_TOK * N_TOK) return;
    int j = idx % N_TOK;
    int i = (idx / N_TOK) % N_TOK;
    int h = idx / (N_TOK * N_TOK);
    float d0 = (float)(j / 7 - i / 7);
    float d1 = (float)(j % 7 - i % 7);
    float f0 = (d0 == 0.f) ? 0.f : copysignf(log1pf(fabsf(d0)), d0);
    float f1 = (d1 == 0.f) ? 0.f : copysignf(log1pf(fabsf(d1)), d1);
    float acc = b2[h];
    #pragma unroll 4
    for (int o = 0; o < 32; ++o) {
        float hh = fmaxf(0.f, w1[o * 2 + 0] * f0 + w1[o * 2 + 1] * f1 + b1[o]);
        acc += w2[h * 32 + o] * hh;
    }
    bias[idx] = acc;
}

// f32 -> bf16 weight conversion (4 elements / thread)
__global__ void cvt_bf16_kernel(const float* __restrict__ src,
                                unsigned short* __restrict__ dst, int n4) {
    int i = blockIdx.x * 256 + threadIdx.x;
    if (i >= n4) return;
    float4 f = ((const float4*)src)[i];
    uint2 r; r.x = pack2(f.x, f.y); r.y = pack2(f.z, f.w);
    ((uint2*)dst)[i] = r;
}

// ---------------------------------------------------------------------------
// Fragment loaders: contiguous per-lane runs, always 16B aligned.
//   A (16x32 bf16, row-per-lane): two 16B chunks at half*16 and half*16+8
//   B via B^T source [n][K]:      one 32B run at n*ldk + kb + half*16
// ---------------------------------------------------------------------------
__device__ __forceinline__ Frag loadA(const unsigned short* rp, int half) {
    Frag f;
    f.q[0] = *(const uint4*)(rp + half * 16);
    f.q[1] = *(const uint4*)(rp + half * 16 + 8);
    return f;
}
__device__ __forceinline__ Frag loadBT(const unsigned short* bp) {
    Frag f;
    f.q[0] = *(const uint4*)(bp);
    f.q[1] = *(const uint4*)(bp + 8);
    return f;
}

// ---------------------------------------------------------------------------
__global__ __launch_bounds__(256)
void winattn_kernel(const float* __restrict__ x, const float* __restrict__ mask,
                    const unsigned short* __restrict__ qkvw,   // bf16 [1152][384]
                    const float* __restrict__ qkv_b,
                    const unsigned short* __restrict__ projw,  // bf16 [384][384]
                    const float* __restrict__ proj_b,
                    const float* __restrict__ posb, float* __restrict__ out) {
    extern __shared__ __align__(16) unsigned char smem[];
    unsigned short* xb = (unsigned short*)(smem + XB_OFF);
    unsigned short* qh = (unsigned short*)(smem + QH_OFF);
    unsigned short* kh = (unsigned short*)(smem + KH_OFF);
    unsigned short* vT = (unsigned short*)(smem + VT_OFF);
    unsigned short* at = (unsigned short*)(smem + AT_OFF);
    unsigned short* os = (unsigned short*)(smem + OS_OFF);

    const int b    = blockIdx.x;
    const int tid  = threadIdx.x;
    const int wave = tid >> 5;
    const int lane = tid & 31;
    const int half = lane >> 4;
    const int ln   = lane & 15;
    const int wb   = b & (NMASK - 1);

    // ---- stage x window into LDS as bf16; zero pad rows of xb and os ----
    const float* xg = x + (size_t)b * (N_TOK * DIM);
    for (int i = tid; i < (N_TOK * DIM) / 4; i += 256) {
        float4 f = ((const float4*)xg)[i];
        uint2 r; r.x = pack2(f.x, f.y); r.y = pack2(f.z, f.w);
        ((uint2*)xb)[i] = r;
    }
    {
        uint4 z; z.x = z.y = z.z = z.w = 0u;
        uint4* xpad = (uint4*)(xb + N_TOK * DIM);   // 15*384 bf16 = 720 uint4
        uint4* opad = (uint4*)(os + N_TOK * DIM);
        for (int i = tid; i < 720; i += 256) { xpad[i] = z; opad[i] = z; }
    }
    __syncthreads();

    // ================= per head-pair loop =================
    for (int hp = 0; hp < NHEADS / 2; ++hp) {
        const int h0 = hp * 2;

        // ---- phase A: qkv GEMM for this head pair.
        // Wave owns one row-tile (mtile = wave>>1) and 6 column tiles.
        {
            const int mtile  = wave >> 1;
            const int ntbase = (wave & 1) * 6;
            const int m      = mtile * 16 + ln;
            v8f acc[6];
            #pragma unroll
            for (int t = 0; t < 6; ++t) acc[t] = (v8f){0.f,0.f,0.f,0.f,0.f,0.f,0.f,0.f};
            for (int ks = 0; ks < DIM / 32; ++ks) {
                Frag a = loadA(xb + m * DIM + ks * 32, half);
                #pragma unroll
                for (int t = 0; t < 6; ++t) {
                    int nt  = ntbase + t;
                    int sel = nt >> 2, wn = nt & 3;
                    int hl  = wn >> 1, ddb = (wn & 1) * 16;
                    int oc  = sel * DIM + (h0 + hl) * HEADD + ddb + ln;
                    Frag bw = loadBT(qkvw + (size_t)oc * DIM + ks * 32 + half * 16);
                    acc[t] = __builtin_amdgcn_wmma_f32_16x16x32_bf16(
                                 false, a.v, false, bw.v, (short)0, acc[t], false, false);
                }
            }
            #pragma unroll
            for (int t = 0; t < 6; ++t) {
                int nt  = ntbase + t;
                int sel = nt >> 2, wn = nt & 3;
                int hl  = wn >> 1, ddb = (wn & 1) * 16;
                int oc  = sel * DIM + (h0 + hl) * HEADD + ddb + ln;
                float bias = qkv_b[oc];
                #pragma unroll
                for (int r = 0; r < 8; ++r) {
                    int row = mtile * 16 + r + half * 8;
                    float val = acc[t][r] + bias;
                    if (sel == 0) {
                        qh[hl * 2048 + row * 32 + ddb + ln] = f2bf(val * QSCALE);
                    } else if (sel == 1) {
                        kh[hl * 2048 + row * 32 + ddb + ln] = f2bf(val);
                    } else {
                        if (row >= N_TOK) val = 0.f;   // zero K-padding for AV
                        vT[hl * 2048 + (ddb + ln) * 64 + row] = f2bf(val);
                    }
                }
            }
        }
        __syncthreads();

        // ---- phase B: one (head, row-tile) per wave: QK^T + softmax + AV ----
        {
            int hl = wave >> 2, mtile = wave & 3;
            int h = h0 + hl, mbase = mtile * 16;

            Frag aq = loadA(qh + hl * 2048 + (mbase + ln) * 32, half);
            float lg[4][8];
            #pragma unroll
            for (int nt = 0; nt < 4; ++nt) {
                Frag bk = loadBT(kh + hl * 2048 + (nt * 16 + ln) * 32 + half * 16);
                v8f acc = {0.f,0.f,0.f,0.f,0.f,0.f,0.f,0.f};
                acc = __builtin_amdgcn_wmma_f32_16x16x32_bf16(
                          false, aq.v, false, bk.v, (short)0, acc, false, false);
                #pragma unroll
                for (int r = 0; r < 8; ++r) {
                    int row = mbase + r + half * 8, col = nt * 16 + ln;
                    float l = acc[r];
                    if (col < N_TOK) {
                        if (row < N_TOK)
                            l += posb[h * 2401 + row * 49 + col]
                               + mask[wb * 2401 + row * 49 + col];
                    } else {
                        l = -1e30f;                    // pad columns -> zero weight
                    }
                    lg[nt][r] = l;
                }
            }
            // row-wise softmax (row spread over 16 lanes of one half)
            #pragma unroll
            for (int r = 0; r < 8; ++r) {
                float mx = fmaxf(fmaxf(lg[0][r], lg[1][r]), fmaxf(lg[2][r], lg[3][r]));
                #pragma unroll
                for (int off = 8; off >= 1; off >>= 1)
                    mx = fmaxf(mx, __shfl_xor(mx, off, 32));
                float s = 0.f;
                #pragma unroll
                for (int nt = 0; nt < 4; ++nt) {
                    float e = __expf(lg[nt][r] - mx);
                    lg[nt][r] = e; s += e;
                }
                #pragma unroll
                for (int off = 8; off >= 1; off >>= 1)
                    s += __shfl_xor(s, off, 32);
                float inv = 1.0f / s;
                #pragma unroll
                for (int nt = 0; nt < 4; ++nt)
                    at[wave * 1024 + (r + half * 8) * 64 + nt * 16 + ln]
                        = f2bf(lg[nt][r] * inv);
            }
            // AV: [16x64] x [64x32]
            #pragma unroll
            for (int nt2 = 0; nt2 < 2; ++nt2) {
                v8f acc = {0.f,0.f,0.f,0.f,0.f,0.f,0.f,0.f};
                #pragma unroll
                for (int ks = 0; ks < 2; ++ks) {
                    Frag a  = loadA(at + wave * 1024 + ln * 64 + ks * 32, half);
                    Frag bv = loadBT(vT + hl * 2048 + (nt2 * 16 + ln) * 64
                                        + ks * 32 + half * 16);
                    acc = __builtin_amdgcn_wmma_f32_16x16x32_bf16(
                              false, a.v, false, bv.v, (short)0, acc, false, false);
                }
                #pragma unroll
                for (int r = 0; r < 8; ++r) {
                    int row = mbase + r + half * 8;
                    if (row < N_TOK)
                        os[row * DIM + h * HEADD + nt2 * 16 + ln] = f2bf(acc[r]);
                }
            }
        }
        __syncthreads();
    }

    // ---- phase C: output projection  y = os @ proj_w^T + proj_b ----
    float* og = out + (size_t)b * (N_TOK * DIM);
    {
        const int mtile = wave >> 1;
        const int m     = mtile * 16 + ln;
        for (int pass = 0; pass < 2; ++pass) {
            const int nb = (wave & 1) * 12 + pass * 6;  // first of 6 column tiles
            v8f acc[6];
            #pragma unroll
            for (int t = 0; t < 6; ++t) acc[t] = (v8f){0.f,0.f,0.f,0.f,0.f,0.f,0.f,0.f};
            for (int ks = 0; ks < DIM / 32; ++ks) {
                Frag a = loadA(os + m * DIM + ks * 32, half);
                #pragma unroll
                for (int t = 0; t < 6; ++t) {
                    int oc = (nb + t) * 16 + ln;
                    Frag bw = loadBT(projw + (size_t)oc * DIM + ks * 32 + half * 16);
                    acc[t] = __builtin_amdgcn_wmma_f32_16x16x32_bf16(
                                 false, a.v, false, bw.v, (short)0, acc[t], false, false);
                }
            }
            #pragma unroll
            for (int t = 0; t < 6; ++t) {
                int oc = (nb + t) * 16 + ln;
                float pb = proj_b[oc];
                #pragma unroll
                for (int r = 0; r < 8; ++r) {
                    int row = mtile * 16 + r + half * 8;
                    if (row < N_TOK) og[row * DIM + oc] = acc[t][r] + pb;
                }
            }
        }
    }
}

// ---------------------------------------------------------------------------
extern "C" void kernel_launch(void* const* d_in, const int* in_sizes, int n_in,
                              void* d_out, int out_size, void* d_ws, size_t ws_size,
                              hipStream_t stream) {
    const float* x      = (const float*)d_in[0];
    const float* mask   = (const float*)d_in[1];
    const float* qkv_w  = (const float*)d_in[2];
    const float* qkv_b  = (const float*)d_in[3];
    const float* proj_w = (const float*)d_in[4];
    const float* proj_b = (const float*)d_in[5];
    const float* pm_w1  = (const float*)d_in[6];
    const float* pm_b1  = (const float*)d_in[7];
    const float* pm_w2  = (const float*)d_in[8];
    const float* pm_b2  = (const float*)d_in[9];
    float* outp = (float*)d_out;

    float*          posb     = (float*)d_ws;
    unsigned short* qkvw_bf  = (unsigned short*)((char*)d_ws + WS_QKVW_OFF);
    unsigned short* projw_bf = (unsigned short*)((char*)d_ws + WS_PROJW_OFF);

    (void)in_sizes; (void)n_in; (void)out_size; (void)ws_size;

    int nbias = NHEADS * N_TOK * N_TOK;
    posbias_kernel<<<(nbias + 255) / 256, 256, 0, stream>>>(pm_w1, pm_b1, pm_w2, pm_b2, posb);

    int n4q = (3 * DIM * DIM) / 4;   // 110592
    int n4p = (DIM * DIM) / 4;       //  36864
    cvt_bf16_kernel<<<(n4q + 255) / 256, 256, 0, stream>>>(qkv_w, qkvw_bf, n4q);
    cvt_bf16_kernel<<<(n4p + 255) / 256, 256, 0, stream>>>(proj_w, projw_bf, n4p);

    hipFuncSetAttribute((const void*)winattn_kernel,
                        hipFuncAttributeMaxDynamicSharedMemorySize, SMEM_BYTES);
    winattn_kernel<<<NWIN, 256, SMEM_BYTES, stream>>>(
        x, mask, qkvw_bf, qkv_b, projw_bf, proj_b, posb, outp);
}
